// QuantizedLinear_2362232013179
// MI455X (gfx1250) — matrix operand order repaired and checked by reading, not compiled, not run on hardware
//
#include <hip/hip_runtime.h>
#include <hip/hip_bf16.h>

typedef __attribute__((ext_vector_type(16))) __bf16 v16bf;
typedef __attribute__((ext_vector_type(8)))  __bf16 v8bf;
typedef __attribute__((ext_vector_type(4)))  __bf16 v4bf;
typedef __attribute__((ext_vector_type(8)))  float  v8f;
typedef __attribute__((ext_vector_type(4)))  float  v4f;
typedef __attribute__((ext_vector_type(4)))  int    v4i;

#define TILE_M 128
#define TILE_N 128
#define TILE_K 32
#define LDS_STRIDE 40   // 32 + 8 pad: 80B row stride (20 dwords) -> conflict-light ds_load_b128

constexpr int IN_DIM  = 4096;
constexpr int OUT_DIM = 16384;
constexpr int M_DIM   = 4 * 2048;   // B * S

// Split barrier with DS-only drain: unlike __syncthreads() (which emits
// s_wait_loadcnt_dscnt 0x0 and kills the global prefetch pipeline), this only
// requires our LDS ops to retire; in-flight global_load_b128 prefetches stay
// outstanding across the barrier and are covered by register-use waits.
__device__ __forceinline__ void wg_barrier_ds_only() {
    asm volatile(
        "s_wait_dscnt 0x0\n\t"
        "s_barrier_signal -1\n\t"
        "s_barrier_wait -1"
        ::: "memory");
}

// ---------------------------------------------------------------------------
// Pass 1: rowsum[r] = sum_k x[r][k]  (enables zp-free dequant in the GEMM loop)
// ---------------------------------------------------------------------------
__global__ __launch_bounds__(256)
void rowsum_kernel(const float* __restrict__ x, float* __restrict__ rs)
{
    const int lane = threadIdx.x & 31;
    const int wid  = threadIdx.x >> 5;
    const int row  = blockIdx.x * 8 + wid;

    const float* p = x + (size_t)row * IN_DIM;
    float s = 0.0f;
#pragma unroll 8
    for (int i = 0; i < IN_DIM / 128; ++i) {          // 32 x float4 per lane
        v4f v = *(const v4f*)(p + i * 128 + lane * 4);
        s += v.x + v.y + v.z + v.w;
    }
#pragma unroll
    for (int off = 16; off > 0; off >>= 1)            // wave32 tree reduce
        s += __shfl_down(s, off, 32);
    if (lane == 0) rs[row] = s;
}

// ---------------------------------------------------------------------------
// Pass 2: out = scale * (bf16(x) @ bf16(q)^T - zp * rowsum(x))
// 128x128 block tile, 8 waves, each wave 32x64 = 2x4 WMMA fragments.
// Double-buffered LDS, one DS-only barrier per k-step, register-staged prefetch
// that now stays in flight across the barrier.
// ---------------------------------------------------------------------------
__global__ __launch_bounds__(256)
void qlinear_wmma_bf16_kernel(const float* __restrict__ x,
                              const int*   __restrict__ qw,
                              const float* __restrict__ scale,
                              const int*   __restrict__ zp,
                              const float* __restrict__ rsum,
                              float*       __restrict__ out)
{
    __shared__ __align__(16) __bf16 sx[2][TILE_M * LDS_STRIDE];  // activations
    __shared__ __align__(16) __bf16 sw[2][TILE_N * LDS_STRIDE];  // weights (q as bf16)

    const int tid  = threadIdx.x;
    const int lane = tid & 31;
    const int wid  = tid >> 5;      // 0..7
    const int half = lane >> 4;     // 0 | 1
    const int l16  = lane & 15;

    const int mTile = blockIdx.x * TILE_M;   // M fastest: weight tile stays L2-resident
    const int nTile = blockIdx.y * TILE_N;

    // global staging: each thread 4x float4 + 4x int4 per k-step.
    // 32-bit voffset + 24-bit ioffset addressing (row offsets i*512KB fold into ioffset).
    const int ldRow = tid >> 3;          // 0..31
    const int ldCol = (tid & 7) * 4;     // 0..28
    const int xoff  = ldRow * IN_DIM + ldCol;
    const float* xbase = x  + (size_t)mTile * IN_DIM;
    const int*   qbase = qw + (size_t)nTile * IN_DIM;

    const int waveM = (wid & 3) * 32;
    const int waveN = (wid >> 2) * 64;

    v8f acc[2][4] = {};
    v4f gx[4];
    v4i gq[4];

    auto loadRegs = [&](int ktn) {
        const int kOff = ktn * TILE_K;   // uniform -> scalar add
#pragma unroll
        for (int i = 0; i < 4; ++i) {
            gx[i] = *(const v4f*)(xbase + kOff + xoff + i * 32 * IN_DIM);
            gq[i] = *(const v4i*)(qbase + kOff + xoff + i * 32 * IN_DIM);
        }
    };

    auto storeTile = [&](int b) {
#pragma unroll
        for (int i = 0; i < 4; ++i) {
            const int r = ldRow + 32 * i;
            v4f fx = gx[i];
            *(v4bf*)(&sx[b][r * LDS_STRIDE + ldCol]) =
                (v4bf){ (__bf16)fx.x, (__bf16)fx.y, (__bf16)fx.z, (__bf16)fx.w };
            v4i qi = gq[i];   // q in 0..15: exact in bf16, zp handled in epilogue
            *(v4bf*)(&sw[b][r * LDS_STRIDE + ldCol]) =
                (v4bf){ (__bf16)(float)qi.x, (__bf16)(float)qi.y,
                        (__bf16)(float)qi.z, (__bf16)(float)qi.w };
        }
    };

    // prologue: tile 0 -> LDS buf 0; prefetch tile 1 into registers
    loadRegs(0);
    storeTile(0);
    loadRegs(1);
    wg_barrier_ds_only();

    const int KT = IN_DIM / TILE_K;   // 128
    for (int kt = 0; kt < KT; ++kt) {
        const int cur = kt & 1;

        // A fragments (16-bit 16x32): lanes 0-15 K={0..7,16..23}, lanes 16-31 K={8..15,24..31}
        v16bf afrag[2];
#pragma unroll
        for (int m = 0; m < 2; ++m) {
            const int row = waveM + m * 16 + l16;
            const int k0  = half * 8;
            v8bf lo = *(const v8bf*)(&sx[cur][row * LDS_STRIDE + k0]);
            v8bf hi = *(const v8bf*)(&sx[cur][row * LDS_STRIDE + k0 + 16]);
            afrag[m] = __builtin_shufflevector(lo, hi, 0,1,2,3,4,5,6,7,8,9,10,11,12,13,14,15);
        }
        // B fragments (32x16): lane holds column N=l16; lanes 0-15 K={0..7,16..23}, lanes 16-31 K={8..15,24..31}
        v16bf bfrag[4];
#pragma unroll
        for (int n = 0; n < 4; ++n) {
            const int row = waveN + n * 16 + l16;
            const int kb  = half * 8;
            v8bf lo = *(const v8bf*)(&sw[cur][row * LDS_STRIDE + kb]);
            v8bf hi = *(const v8bf*)(&sw[cur][row * LDS_STRIDE + kb + 16]);
            bfrag[n] = __builtin_shufflevector(lo, hi, 0,1,2,3,4,5,6,7,8,9,10,11,12,13,14,15);
        }

        // write NEXT tile into the alternate buffer (no conflict with current reads)
        if (kt + 1 < KT) storeTile(cur ^ 1);
        // prefetch tile kt+2 from global; stays in flight across the barrier
        if (kt + 2 < KT) loadRegs(kt + 2);

        // 8 WMMAs per wave per k-step
#pragma unroll
        for (int m = 0; m < 2; ++m)
#pragma unroll
            for (int n = 0; n < 4; ++n)
                acc[m][n] = __builtin_amdgcn_wmma_f32_16x16x32_bf16(
                    false, afrag[m], false, bfrag[n],
                    (short)0, acc[m][n], false, false);

        wg_barrier_ds_only();   // drains DScnt only; prefetch loads remain outstanding
    }

    // epilogue. C layout: VGPR v -> M = v + 8*half; N = l16.
    // out = acc*scale[col] - rowsum[row]*(scale[col]*zp[col])
    float rs[2][8];
#pragma unroll
    for (int m = 0; m < 2; ++m) {
        const int rowBase = mTile + waveM + m * 16 + half * 8;
#pragma unroll
        for (int v = 0; v < 8; ++v)
            rs[m][v] = rsum[rowBase + v];
    }

#pragma unroll
    for (int n = 0; n < 4; ++n) {
        const int col = nTile + waveN + n * 16 + l16;
        const float s1 = scale[col];
        const float s2 = s1 * (float)zp[col];
#pragma unroll
        for (int m = 0; m < 2; ++m) {
            const int rowBase = mTile + waveM + m * 16 + half * 8;
#pragma unroll
            for (int v = 0; v < 8; ++v) {
                const float val = acc[m][n][v] * s1 - rs[m][v] * s2;
                // NT store: don't evict L2-resident x with the 536MB output stream
                __builtin_nontemporal_store(val, out + (size_t)(rowBase + v) * OUT_DIM + col);
            }
        }
    }
}

extern "C" void kernel_launch(void* const* d_in, const int* in_sizes, int n_in,
                              void* d_out, int out_size, void* d_ws, size_t ws_size,
                              hipStream_t stream) {
    const float* x     = (const float*)d_in[0];
    const int*   qw    = (const int*)d_in[1];
    const float* scale = (const float*)d_in[2];
    const int*   zp    = (const int*)d_in[3];
    float*       out   = (float*)d_out;
    float*       rsum  = (float*)d_ws;          // 8192 floats = 32KB scratch

    // Pass 1: row sums of x
    rowsum_kernel<<<M_DIM / 8, 256, 0, stream>>>(x, rsum);

    // Pass 2: WMMA GEMM (M fastest -> weight tiles reused from L2, weights stream HBM once)
    dim3 grid(M_DIM / TILE_M, OUT_DIM / TILE_N);   // 64 x 128
    qlinear_wmma_bf16_kernel<<<grid, 256, 0, stream>>>(x, qw, scale, zp, rsum, out);
}
